// SparseLinear_12713103196329
// MI455X (gfx1250) — matrix operand order repaired and checked by reading, not compiled, not run on hardware
//
#include <hip/hip_runtime.h>

// ---------------------------------------------------------------------------
// SparseLinear on MI455X (gfx1250):
//   Fixed unstructured 10% sparsity -> densify W to bf16 once, then dense
//   bf16 WMMA GEMM (f32 accumulate):  out = x @ W^T + bias.
//   GEMM stages tiles in LDS with the Tensor Data Mover (tensor_load_to_lds,
//   TENSORcnt, hardware LDS pad for bank-conflict-free rows), double-buffered.
//   The TDM's LDS writes are invisible to the compiler; we escape the smem
//   pointer into a memory-clobber asm each stage so the ds_loads cannot be
//   folded, while still lowering as plain ds_load_b128.
// ---------------------------------------------------------------------------

typedef __attribute__((ext_vector_type(16))) __bf16 v16bf;
typedef __attribute__((ext_vector_type(8)))  __bf16 v8bf;
typedef __attribute__((ext_vector_type(8)))  float  v8f;
typedef __attribute__((ext_vector_type(4)))  unsigned int u32x4v;
typedef __attribute__((ext_vector_type(8)))  int i32x8v;
typedef __attribute__((ext_vector_type(4)))  int i32x4v;

#define IN_CH  4096
#define OUT_CH 4096
#define BATCH  8192

// GEMM tiling
#define KB          32                  // K per stage
#define ROW_DATA_B  (KB * 2)            // 64B of bf16 per tile row
#define ROW_PAD_B   16                  // TDM-inserted pad (4 dwords)
#define ROW_STRIDE  (ROW_DATA_B + ROW_PAD_B)   // 80B -> conflict-free banks
#define TILE_B      (128 * ROW_STRIDE)  // 10240B per 128x32 tile
#define STAGE_B     (2 * TILE_B)        // A + B = 20480B
#define NSTAGE      (IN_CH / KB)        // 128

// ---- zero the dense bf16 weight buffer (16B stores) -----------------------
__global__ void sl_zero(u32x4v* __restrict__ p, int n16) {
    int i = blockIdx.x * blockDim.x + threadIdx.x;
    if (i < n16) {
        u32x4v z = {0u, 0u, 0u, 0u};
        p[i] = z;
    }
}

// ---- scatter COO nnz into dense bf16 W[out_ch][in_ch] ---------------------
__global__ void sl_scatter(const float* __restrict__ w,
                           const int* __restrict__ rows,
                           const int* __restrict__ cols,
                           __bf16* __restrict__ Wd, int nnz) {
    int i = blockIdx.x * blockDim.x + threadIdx.x;
    if (i < nnz)
        Wd[(size_t)rows[i] * IN_CH + cols[i]] = (__bf16)w[i];
}

// ---- convert x f32 -> bf16 (8 elems/thread, 16B stores) -------------------
__global__ void sl_cvt(const float* __restrict__ x,
                       __bf16* __restrict__ xb, int n8) {
    int i = blockIdx.x * blockDim.x + threadIdx.x;
    if (i < n8) {
        const float4* xv = (const float4*)x;
        float4 a = xv[2 * i];
        float4 b = xv[2 * i + 1];
        v8bf o;
        o[0] = (__bf16)a.x; o[1] = (__bf16)a.y;
        o[2] = (__bf16)a.z; o[3] = (__bf16)a.w;
        o[4] = (__bf16)b.x; o[5] = (__bf16)b.y;
        o[6] = (__bf16)b.z; o[7] = (__bf16)b.w;
        ((v8bf*)xb)[i] = o;
    }
}

// ---- TDM: 2D tile (tile_d0 x tile_d1 elements) global -> LDS --------------
// Descriptor packing per CDNA5 ISA 8.3/8.4. data_size=2B. Hardware pads LDS
// destination: 4 dwords after every 16 dwords (64B row -> 80B stride).
__device__ __forceinline__ void tdm_load_tile_2d(
    unsigned int lds_byte_off, const void* gptr,
    unsigned int tensor_d0, unsigned int tensor_d1,
    unsigned int tile_d0, unsigned int tile_d1,
    unsigned int stride0_elems)
{
    unsigned long long ga = (unsigned long long)gptr;
    u32x4v g0;
    g0[0] = 1u;                                           // count=1 (valid)
    g0[1] = lds_byte_off;                                 // lds_addr
    g0[2] = (unsigned int)(ga & 0xFFFFFFFFull);           // global_addr[31:0]
    g0[3] = (unsigned int)((ga >> 32) & 0x01FFFFFFull)    // global_addr[56:32]
          | (2u << 30);                                   // type=2 ("image")
    i32x8v g1;
    g1[0] = (int)((1u << 16)      // data_size = 2 bytes
                | (1u << 20)      // pad_enable
                | (3u << 22)      // pad_interval: 16 dwords
                | (3u << 25));    // pad_amount:   4 dwords
    g1[1] = (int)((tensor_d0 & 0xFFFFu) << 16);                        // dim0 lo
    g1[2] = (int)((tensor_d0 >> 16) | ((tensor_d1 & 0xFFFFu) << 16));  // dim0 hi | dim1 lo
    g1[3] = (int)((tensor_d1 >> 16) | (tile_d0 << 16));                // dim1 hi | tile_dim0
    g1[4] = (int)(tile_d1 & 0xFFFFu);                                  // tile_dim1 (tile_dim2=0)
    g1[5] = (int)stride0_elems;                                        // dim0_stride lo32
    g1[6] = 0;
    g1[7] = 0;
    i32x4v gz4 = {0, 0, 0, 0};                            // groups 2/3: NULL (2D)
    i32x8v gz8 = {0, 0, 0, 0, 0, 0, 0, 0};
    __builtin_amdgcn_tensor_load_to_lds(g0, g1, gz4, gz4, gz8, 0);
}

// ---- WMMA GEMM: out[M,N] = Xbf[M,K] @ Wd[N,K]^T + bias --------------------
// 256 threads = 8 waves (2x4); wave tile 64x32 (4x2 WMMA tiles); workgroup
// tile 128x128; K stepped by 32; TDM double-buffered LDS staging.
__global__ __launch_bounds__(256)
void sl_gemm(const __bf16* __restrict__ Xbf,
             const __bf16* __restrict__ Wd,
             const float* __restrict__ bias,
             float* __restrict__ out) {
    __shared__ unsigned char smem[2 * STAGE_B];           // 40KB

    const int lane = threadIdx.x & 31;
    const int wave = threadIdx.x >> 5;
    const int wm = wave >> 2;                             // 0..1
    const int wn = wave & 3;                              // 0..3
    const int m_blk = blockIdx.y * 128;
    const int n_blk = blockIdx.x * 128;

    const int frow = lane & 15;                           // row-in-tile
    const int aoff = ((lane >> 4) << 3) * 2;              // A byte off: 0/16
    const int boff = ((lane >> 4) << 3) * 2;              // B byte off: 0/16

    // Prologue: stage 0 in flight.
    if (wave == 0) {
        tdm_load_tile_2d(0,      Xbf + (size_t)m_blk * IN_CH, IN_CH, BATCH,  KB, 128, IN_CH);
        tdm_load_tile_2d(TILE_B, Wd  + (size_t)n_blk * IN_CH, IN_CH, OUT_CH, KB, 128, IN_CH);
    }

    v8f acc[4][2] = {};

    for (int s = 0; s < NSTAGE; ++s) {
        if (wave == 0) {
            if (s + 1 < NSTAGE) {
                const int k0 = (s + 1) * KB;
                const unsigned int nb = (unsigned int)(((s + 1) & 1) * STAGE_B);
                tdm_load_tile_2d(nb,          Xbf + (size_t)m_blk * IN_CH + k0, IN_CH, BATCH,  KB, 128, IN_CH);
                tdm_load_tile_2d(nb + TILE_B, Wd  + (size_t)n_blk * IN_CH + k0, IN_CH, OUT_CH, KB, 128, IN_CH);
                __builtin_amdgcn_s_wait_tensorcnt(2);     // stage s complete
            } else {
                __builtin_amdgcn_s_wait_tensorcnt(0);
            }
        }
        __syncthreads();
        // Escape smem into a memory-clobber asm: the TDM engine's LDS writes
        // are invisible to the compiler; this forbids folding/CSE of the
        // ds_loads below while keeping them ordinary addrspace(3) loads.
        asm volatile("" : : "r"(smem) : "memory");

        const unsigned char* As = smem + (s & 1) * STAGE_B;
        const unsigned char* Bs = As + TILE_B;

        // B fragments: K {kh..kh+7} and K {16+kh..16+kh+7} chunks within an 80B row.
        v16bf bfrag[2];
#pragma unroll
        for (int nt = 0; nt < 2; ++nt) {
            const unsigned char* p = Bs + (wn * 32 + nt * 16 + frow) * ROW_STRIDE + boff;
            v8bf lo = *(const v8bf*)p;
            v8bf hi = *(const v8bf*)(p + 32);
            bfrag[nt] = __builtin_shufflevector(
                lo, hi, 0, 1, 2, 3, 4, 5, 6, 7, 8, 9, 10, 11, 12, 13, 14, 15);
        }
#pragma unroll
        for (int mt = 0; mt < 4; ++mt) {
            // A fragment: K {kh..kh+7} and K {16+kh..16+kh+7} chunks.
            const unsigned char* p = As + (wm * 64 + mt * 16 + frow) * ROW_STRIDE + aoff;
            v8bf lo = *(const v8bf*)p;
            v8bf hi = *(const v8bf*)(p + 32);
            v16bf afrag = __builtin_shufflevector(
                lo, hi, 0, 1, 2, 3, 4, 5, 6, 7, 8, 9, 10, 11, 12, 13, 14, 15);
#pragma unroll
            for (int nt = 0; nt < 2; ++nt)
                acc[mt][nt] = __builtin_amdgcn_wmma_f32_16x16x32_bf16(
                    false, afrag, false, bfrag[nt],
                    (short)0, acc[mt][nt], false, false);
        }
        __syncthreads();                                  // buffer reuse guard
    }

    // Epilogue: C/D layout -> VGPR v = M row v (+8 upper half-wave), N=lane&15.
    const int cn   = lane & 15;
    const int moff = (lane >> 4) * 8;
#pragma unroll
    for (int nt = 0; nt < 2; ++nt) {
        const int n = n_blk + wn * 32 + nt * 16 + cn;
        const float bv = bias[n];
#pragma unroll
        for (int mt = 0; mt < 4; ++mt) {
            const int mbase = m_blk + wm * 64 + mt * 16 + moff;
#pragma unroll
            for (int v = 0; v < 8; ++v)
                out[(size_t)(mbase + v) * OUT_CH + n] = acc[mt][nt][v] + bv;
        }
    }
}

// ---------------------------------------------------------------------------
extern "C" void kernel_launch(void* const* d_in, const int* in_sizes, int n_in,
                              void* d_out, int out_size, void* d_ws, size_t ws_size,
                              hipStream_t stream) {
    const float* x    = (const float*)d_in[0];
    const float* sw   = (const float*)d_in[1];
    const float* bias = (const float*)d_in[2];
    const int*   rows = (const int*)d_in[3];
    const int*   cols = (const int*)d_in[4];
    const int    nnz  = in_sizes[1];
    float* out = (float*)d_out;

    char* ws = (char*)d_ws;
    __bf16* Wd = (__bf16*)ws;                                  // 32 MB
    __bf16* Xb = (__bf16*)(ws + (size_t)OUT_CH * IN_CH * 2);   // 64 MB

    // 1) zero dense bf16 W
    int n16 = (OUT_CH * IN_CH) / 8;
    sl_zero<<<(n16 + 255) / 256, 256, 0, stream>>>((u32x4v*)Wd, n16);

    // 2) scatter nnz values
    sl_scatter<<<(nnz + 255) / 256, 256, 0, stream>>>(sw, rows, cols, Wd, nnz);

    // 3) x f32 -> bf16
    int n8 = (BATCH * IN_CH) / 8;
    sl_cvt<<<(n8 + 255) / 256, 256, 0, stream>>>(x, Xb, n8);

    // 4) TDM-staged WMMA GEMM + bias
    dim3 grid(OUT_CH / 128, BATCH / 128);
    sl_gemm<<<grid, 256, 0, stream>>>(Xb, Wd, bias, out);
}
